// elmo_ncrf_recipes_70016556859464
// MI455X (gfx1250) — compile-verified, matrix operand order checked by reading, not run on hardware
//
#include <hip/hip_runtime.h>
#include <hip/hip_bf16.h>
#include <stdint.h>

// ---------------------------------------------------------------------------
// Problem dims (match reference setup_inputs)
// ---------------------------------------------------------------------------
#define T_SEQ   8192
#define D_IN    1024
#define HD      512          // sentence per-direction hidden
#define G4      2048         // 4*HD
#define S_SEG   64
#define H2D     256          // entity per-direction hidden
#define G4E     1024         // 4*H2D

// sentence recurrence partitioning
#define NWG_DIR 16           // workgroups per direction
#define UPW     (HD / NWG_DIR)   // 32 units per WG
#define ROWS_PW (4 * UPW)        // 128 gate rows per WG
#define THR_R   ROWS_PW          // 128 threads per WG

typedef __attribute__((ext_vector_type(16))) __bf16 v16bf;
typedef __attribute__((ext_vector_type(8)))  float  v8f;
typedef __attribute__((ext_vector_type(4)))  unsigned v4u;
typedef __attribute__((ext_vector_type(8)))  int    v8i;
typedef __attribute__((ext_vector_type(4)))  int    v4i;

// Toolchain arity probe: the therock-10.0 headers (6-arg TDM builtin) ship the
// gfx1250 TDM header; ROCm 7.2 (5-arg builtin) does not.
#if __has_include(<hip/amd_detail/amd_gfx1250_TDM.h>)
#define TDM_6ARG 1
#endif

// ---------------------------------------------------------------------------
// Tensor Data Mover: 1-D contiguous copy global -> LDS, nElem 4-byte words.
// Descriptor bitfields per CDNA5 ISA 8.3/8.4 (D# group0/group1).
// ---------------------------------------------------------------------------
__device__ __forceinline__ void tdm_copy_1d(const void* gsrc, void* ldst,
                                            unsigned nElem) {
  // flat shared pointers keep the group-segment byte offset in addr[31:0]
  unsigned ldsOff = (unsigned)(uintptr_t)ldst;
  unsigned long long ga = (unsigned long long)(uintptr_t)gsrc;
  v4u g0;
  g0[0] = 1u;                                       // count=1, user descriptor
  g0[1] = ldsOff;                                   // lds_addr (bytes)
  g0[2] = (unsigned)(ga & 0xFFFFFFFFull);           // global_addr[31:0]
  g0[3] = (unsigned)((ga >> 32) & 0x01FFFFFFull)    // global_addr[56:32]
        | (2u << 30);                               // type=2 ("image")
  v8i g1;
  g1[0] = (int)(2u << 16);                          // workgroup_mask=0, data_size=4B
  g1[1] = (int)((nElem & 0xFFFFu) << 16);           // tensor_dim0[15:0] @ bits63:48
  g1[2] = (int)((nElem >> 16) & 0xFFFFu)            // tensor_dim0[31:16]
        | (1 << 16);                                // tensor_dim1 = 1 (low16)
  g1[3] = (int)((nElem & 0xFFFFu) << 16);           // tile_dim0 @ bits127:112
  g1[4] = 1;                                        // tile_dim1=1, tile_dim2=0
  g1[5] = (int)nElem;                               // tensor_dim0_stride[31:0]
  g1[6] = 0;                                        // stride hi / dim1_stride lo
  g1[7] = 0;
  v4i gz = {0, 0, 0, 0};
#if defined(TDM_6ARG)
  v8i gz8 = {0, 0, 0, 0, 0, 0, 0, 0};
  __builtin_amdgcn_tensor_load_to_lds(g0, g1, gz, gz, gz8, 0);
#else
  __builtin_amdgcn_tensor_load_to_lds(g0, g1, gz, gz, 0);
#endif
}

// ---------------------------------------------------------------------------
// f32 -> bf16 conversion
// ---------------------------------------------------------------------------
__global__ __launch_bounds__(256) void f2bf_kernel(const float* __restrict__ src,
                                                   __bf16* __restrict__ dst, int n) {
  int i = blockIdx.x * 256 + threadIdx.x;
  if (i < n) dst[i] = (__bf16)src[i];
}

// ---------------------------------------------------------------------------
// Init: h double-buffer (parity0 = h0), zero barrier counters
// ---------------------------------------------------------------------------
__global__ __launch_bounds__(1024) void init_kernel(const float* __restrict__ h0_sent,
                                                    float* __restrict__ h_glob,
                                                    int* __restrict__ bar) {
  int i = threadIdx.x;
  if (i < 2 * HD) {
    h_glob[i] = h0_sent[i];        // parity-0 buffer: [dir][unit]
    h_glob[2 * HD + i] = 0.0f;     // parity-1 buffer (written before first read)
  }
  if (i < 2) bar[i] = 0;
}

// ---------------------------------------------------------------------------
// bf16 WMMA GEMM:  C[M,N] = A[M,K] * Bw[N,K]^T   (i.e.  x @ W.T)
// One wave computes a 16x64 strip (4 N-tiles): A fragment loaded once per
// K-step, 4 B fragments, 4 WMMAs -> 10 b128 loads / 4 WMMAs.
// A fragment layout (16-bit A 16x32): lane m=l&15, half=l>>4,
//   VGPR j holds K = 16*(j>>2) + 8*half + 2*(j&3) + {0,1}
// B fragment symmetric (lane = column n).
// D layout: VGPR r, lane l -> M = r + 8*(l>>4), N = l&15.
// ---------------------------------------------------------------------------
__global__ __launch_bounds__(256) void gemm_bf16_wmma(const __bf16* __restrict__ A,
                                                      const __bf16* __restrict__ Bw,
                                                      float* __restrict__ C,
                                                      int M, int N, int K,
                                                      int lda, int ldb, int ldc) {
  const int wave = threadIdx.x >> 5;
  const int lane = threadIdx.x & 31;
  const int n0 = (blockIdx.x * 8 + wave) * 64;   // 4 consecutive N tiles
  const int m0 = blockIdx.y * 16;
  if (m0 >= M || n0 >= N) return;                // wave-uniform

  const int half = lane >> 4;
  const int mr   = lane & 15;
  const __bf16* arow = A + (size_t)(m0 + mr) * lda;
  const __bf16* brow0 = Bw + (size_t)(n0 +  0 + mr) * ldb;
  const __bf16* brow1 = Bw + (size_t)(n0 + 16 + mr) * ldb;
  const __bf16* brow2 = Bw + (size_t)(n0 + 32 + mr) * ldb;
  const __bf16* brow3 = Bw + (size_t)(n0 + 48 + mr) * ldb;

  v8f acc0 = {0.f,0.f,0.f,0.f,0.f,0.f,0.f,0.f};
  v8f acc1 = acc0, acc2 = acc0, acc3 = acc0;

  for (int k0 = 0; k0 < K; k0 += 32) {
    if (k0 + 32 < K) {
      __builtin_prefetch((const void*)(arow  + k0 + 32), 0, 1);
      __builtin_prefetch((const void*)(brow0 + k0 + 32), 0, 1);
    }
    union { v16bf v; unsigned u[8]; } fa, fb0, fb1, fb2, fb3;
#pragma unroll
    for (int j = 0; j < 8; ++j) {
      int kk = k0 + ((j >> 2) << 4) + (half << 3) + ((j & 3) << 1);
      fa.u[j]  = *(const unsigned*)(arow  + kk);   // packed bf16 pair
      fb0.u[j] = *(const unsigned*)(brow0 + kk);
      fb1.u[j] = *(const unsigned*)(brow1 + kk);
      fb2.u[j] = *(const unsigned*)(brow2 + kk);
      fb3.u[j] = *(const unsigned*)(brow3 + kk);
    }
    acc0 = __builtin_amdgcn_wmma_f32_16x16x32_bf16(false, fa.v, false, fb0.v,
                                                   (short)0, acc0, false, false);
    acc1 = __builtin_amdgcn_wmma_f32_16x16x32_bf16(false, fa.v, false, fb1.v,
                                                   (short)0, acc1, false, false);
    acc2 = __builtin_amdgcn_wmma_f32_16x16x32_bf16(false, fa.v, false, fb2.v,
                                                   (short)0, acc2, false, false);
    acc3 = __builtin_amdgcn_wmma_f32_16x16x32_bf16(false, fa.v, false, fb3.v,
                                                   (short)0, acc3, false, false);
  }

#pragma unroll
  for (int r = 0; r < 8; ++r) {
    size_t row = (size_t)(m0 + r + (half << 3)) * ldc;
    C[row + n0 +  0 + mr] = acc0[r];
    C[row + n0 + 16 + mr] = acc1[r];
    C[row + n0 + 32 + mr] = acc2[r];
    C[row + n0 + 48 + mr] = acc3[r];
  }
}

// ---------------------------------------------------------------------------
// Sentence biLSTM recurrence.
// 32 workgroups: WGs 0..15 forward, 16..31 backward. Each WG owns 32 hidden
// units (all 4 gates = 128 rows); its Whh slice is DMA'd into LDS by the
// Tensor Data Mover (4x 64KB 1-D descriptors). h is exchanged via a parity
// double-buffer in global memory and a monotonic atomic-counter barrier.
// ---------------------------------------------------------------------------
__global__ __launch_bounds__(THR_R) void lstm_sent_kernel(
    const float* __restrict__ gx,          // [T, 4096] fwd gates | bwd gates
    const float* __restrict__ Whh_f,       // [2048, 512]
    const float* __restrict__ Whh_b,
    const float* __restrict__ b_f,         // [2048]
    const float* __restrict__ b_b,
    const float* __restrict__ c0_sent,     // [2, 512]
    float* __restrict__ lstm_out,          // [T, 1024]
    float* __restrict__ h_glob,            // [2 parity][2 dir][512]
    int* __restrict__ bar) {               // [2] counters
  extern __shared__ float lds[];
  float* wsh = lds;                          // [128][512] weight slice
  float* hsh = wsh + ROWS_PW * HD;           // [512]
  float* zsh = hsh + HD;                     // [128]

  const int wg   = blockIdx.x;
  const int dir  = wg >> 4;        // 0 fwd, 1 bwd
  const int wloc = wg & 15;
  const int tid  = threadIdx.x;    // 0..127
  const int gate = tid >> 5;       // 0=i 1=f 2=g 3=o
  const int ulc  = tid & 31;
  const int unit = wloc * UPW + ulc;        // 0..511
  const int row  = gate * HD + unit;        // 0..2047

  const float* Whh = dir ? Whh_b : Whh_f;
  const float* bb  = dir ? b_b   : b_f;
  int* cnt = bar + dir;

  // TDM: stage this WG's 4 contiguous gate chunks (32 rows x 512 f32 each)
  if (tid < 32) {                  // wave 0 issues the descriptors
#pragma unroll
    for (int g = 0; g < 4; ++g) {
      const float* src = Whh + ((size_t)g * HD + wloc * UPW) * HD;
      float* dst = wsh + (size_t)g * UPW * HD;
      tdm_copy_1d(src, dst, UPW * HD);     // 16384 dwords
    }
    __builtin_amdgcn_s_wait_tensorcnt(0);
  }
  const float bias = bb[row];
  float c = 0.0f;
  if (tid < UPW) c = c0_sent[dir * HD + wloc * UPW + tid];
  __syncthreads();

  const float* wd = wsh + tid * HD;

  for (int s = 0; s < T_SEQ; ++s) {
    const int tt = dir ? (T_SEQ - 1 - s) : s;

    // gather full h (previous step, parity s&1) into LDS
    const volatile float* hgr = h_glob + ((size_t)((s & 1) * 2 + dir)) * HD;
    for (int k = tid; k < HD; k += THR_R) hsh[k] = hgr[k];
    __syncthreads();

    float acc = bias + gx[(size_t)tt * 4096 + dir * G4 + row];
#pragma unroll 8
    for (int k = 0; k < HD; ++k) acc += wd[k] * hsh[k];
    zsh[tid] = acc;
    __syncthreads();

    if (tid < UPW) {
      float zi = zsh[tid], zf = zsh[32 + tid], zg = zsh[64 + tid], zo = zsh[96 + tid];
      float fi = 1.0f / (1.0f + __expf(-zi));
      float ff = 1.0f / (1.0f + __expf(-zf));
      float fg = tanhf(zg);
      float fo = 1.0f / (1.0f + __expf(-zo));
      c = ff * c + fi * fg;
      float hn = fo * tanhf(c);
      const int u = wloc * UPW + tid;
      ((volatile float*)h_glob)[((size_t)(((s + 1) & 1) * 2 + dir)) * HD + u] = hn;
      lstm_out[(size_t)tt * 1024 + dir * HD + u] = hn;
    }
    __threadfence();
    __syncthreads();

    // per-direction grid barrier (monotonic counter, 16 WGs)
    if (tid == 0) {
      atomicAdd(cnt, 1);
      const int target = (s + 1) * NWG_DIR;
      while (*((volatile int*)cnt) < target) __builtin_amdgcn_s_sleep(1);
    }
    __syncthreads();
  }
}

// ---------------------------------------------------------------------------
// attn0[j] = W_attn[j,:] . elmo_ings[0,:] + b_attn[j]   (only ingredient 0
// is consumed by the reference: w[:, :, 0])
// ---------------------------------------------------------------------------
__global__ __launch_bounds__(256) void attn0_kernel(const float* __restrict__ W_attn,
                                                    const float* __restrict__ b_attn,
                                                    const float* __restrict__ elmo_ings,
                                                    float* __restrict__ attn0) {
  int j = blockIdx.x * 256 + threadIdx.x;
  if (j >= 1024) return;
  const float* wr = W_attn + (size_t)j * 1024;
  float acc = b_attn[j];
#pragma unroll 4
  for (int k = 0; k < 1024; ++k) acc += wr[k] * elmo_ings[k];
  attn0[j] = acc;
}

// ---------------------------------------------------------------------------
// Per-segment: scores = lstm_out . attn0, masked softmax over the segment,
// step_features[s] = sum_t w_t * lstm_out[t].  One block per segment.
// ---------------------------------------------------------------------------
__global__ __launch_bounds__(256) void seg_attn_kernel(const float* __restrict__ lstm_out,
                                                       const float* __restrict__ attn0,
                                                       const int* __restrict__ indices,
                                                       float* __restrict__ scores,
                                                       float* __restrict__ stepf) {
  __shared__ float red[256];
  const int s = blockIdx.x;
  const int tid = threadIdx.x;
  const int lo = indices[s] + 1;
  const int hi = indices[s + 1];           // inclusive

  // pass 1: scores + max
  float m = -3.0e38f;
  for (int t = lo + tid; t <= hi; t += 256) {
    const float* lr = lstm_out + (size_t)t * 1024;
    float sc = 0.0f;
#pragma unroll 4
    for (int k = 0; k < 1024; ++k) sc += lr[k] * attn0[k];
    scores[t] = sc;
    m = fmaxf(m, sc);
  }
  red[tid] = m; __syncthreads();
  for (int o = 128; o > 0; o >>= 1) {
    if (tid < o) red[tid] = fmaxf(red[tid], red[tid + o]);
    __syncthreads();
  }
  m = red[0]; __syncthreads();

  // pass 2: exp + sum
  float sum = 0.0f;
  for (int t = lo + tid; t <= hi; t += 256) {
    float e = __expf(scores[t] - m);
    scores[t] = e;
    sum += e;
  }
  red[tid] = sum; __syncthreads();
  for (int o = 128; o > 0; o >>= 1) {
    if (tid < o) red[tid] += red[tid + o];
    __syncthreads();
  }
  const float inv = 1.0f / red[0]; __syncthreads();

  // pass 3: weighted sum of lstm_out rows
  for (int d = tid; d < 1024; d += 256) {
    float acc = 0.0f;
    for (int t = lo; t <= hi; ++t) acc += scores[t] * lstm_out[(size_t)t * 1024 + d];
    stepf[(size_t)s * 1024 + d] = acc * inv;
  }
}

// ---------------------------------------------------------------------------
// Entity biLSTM recurrence: 2 WGs (one per direction), 1024 threads = one
// gate row each; h[256] and z[1024] in LDS; 64 steps.
// ---------------------------------------------------------------------------
__global__ __launch_bounds__(1024) void lstm_ent_kernel(
    const float* __restrict__ gxe,        // [64, 2048] ef | eb
    const float* __restrict__ Whh_ef,     // [1024, 256]
    const float* __restrict__ Whh_eb,
    const float* __restrict__ b_ef,       // [1024]
    const float* __restrict__ b_eb,
    const float* __restrict__ h0_ent,     // [2, 256]
    const float* __restrict__ c0_ent,
    float* __restrict__ ent) {            // [64, 512]
  __shared__ float hsh[H2D];
  __shared__ float zsh[G4E];
  const int dir = blockIdx.x;
  const int tid = threadIdx.x;            // gate row 0..1023
  const float* Whh = dir ? Whh_eb : Whh_ef;
  const float* bb  = dir ? b_eb   : b_ef;
  const float bias = bb[tid];
  const float* wr = Whh + (size_t)tid * H2D;

  float c = 0.0f;
  if (tid < H2D) { hsh[tid] = h0_ent[dir * H2D + tid]; c = c0_ent[dir * H2D + tid]; }
  __syncthreads();

  for (int s = 0; s < S_SEG; ++s) {
    const int tt = dir ? (S_SEG - 1 - s) : s;
    float acc = bias + gxe[(size_t)tt * 2048 + dir * G4E + tid];
#pragma unroll 8
    for (int k = 0; k < H2D; ++k) acc += wr[k] * hsh[k];
    zsh[tid] = acc;
    __syncthreads();
    if (tid < H2D) {
      float zi = zsh[tid], zf = zsh[H2D + tid], zg = zsh[2 * H2D + tid], zo = zsh[3 * H2D + tid];
      float fi = 1.0f / (1.0f + __expf(-zi));
      float ff = 1.0f / (1.0f + __expf(-zf));
      float fg = tanhf(zg);
      float fo = 1.0f / (1.0f + __expf(-zo));
      c = ff * c + fi * fg;
      float hn = fo * tanhf(c);
      ent[(size_t)tt * 512 + dir * H2D + tid] = hn;
      hsh[tid] = hn;
    }
    __syncthreads();
  }
}

// ---------------------------------------------------------------------------
// logits[s, r] = ent[s,:] . W_log[r,:] + b_log[r]
// ---------------------------------------------------------------------------
__global__ __launch_bounds__(128) void logits_kernel(const float* __restrict__ ent,
                                                     const float* __restrict__ W_log,
                                                     const float* __restrict__ b_log,
                                                     float* __restrict__ out) {
  int i = threadIdx.x;
  if (i >= 128) return;
  int s = i >> 1, r = i & 1;
  float acc = b_log[r];
#pragma unroll 4
  for (int k = 0; k < 512; ++k) acc += ent[(size_t)s * 512 + k] * W_log[r * 512 + k];
  out[s * 2 + r] = acc;
}

// ---------------------------------------------------------------------------
// Host launcher
// ---------------------------------------------------------------------------
static inline size_t align256(size_t x) { return (x + 255) & ~(size_t)255; }

extern "C" void kernel_launch(void* const* d_in, const int* in_sizes, int n_in,
                              void* d_out, int out_size, void* d_ws, size_t ws_size,
                              hipStream_t stream) {
  const float* elmo_sent = (const float*)d_in[0];
  const float* elmo_ings = (const float*)d_in[1];
  const int*   indices   = (const int*)  d_in[2];
  const float* Wih_f  = (const float*)d_in[3];
  const float* Whh_f  = (const float*)d_in[4];
  const float* b_f    = (const float*)d_in[5];
  const float* Wih_b  = (const float*)d_in[6];
  const float* Whh_b  = (const float*)d_in[7];
  const float* b_b    = (const float*)d_in[8];
  const float* h0_sent = (const float*)d_in[9];
  const float* c0_sent = (const float*)d_in[10];
  const float* W_attn = (const float*)d_in[11];
  const float* b_attn = (const float*)d_in[12];
  const float* Wih_ef = (const float*)d_in[13];
  const float* Whh_ef = (const float*)d_in[14];
  const float* b_ef   = (const float*)d_in[15];
  const float* Wih_eb = (const float*)d_in[16];
  const float* Whh_eb = (const float*)d_in[17];
  const float* b_eb   = (const float*)d_in[18];
  const float* h0_ent = (const float*)d_in[19];
  const float* c0_ent = (const float*)d_in[20];
  const float* W_log  = (const float*)d_in[21];
  const float* b_log  = (const float*)d_in[22];
  float* out = (float*)d_out;

  // workspace carve-up (~199 MB)
  uint8_t* ws = (uint8_t*)d_ws;
  size_t off = 0;
  __bf16* xbf   = (__bf16*)(ws + off); off = align256(off + (size_t)T_SEQ * D_IN * 2);
  __bf16* wbf   = (__bf16*)(ws + off); off = align256(off + (size_t)2 * G4 * D_IN * 2);
  float*  gx    = (float*) (ws + off); off = align256(off + (size_t)T_SEQ * 2 * G4 * 4);
  float*  lstm  = (float*) (ws + off); off = align256(off + (size_t)T_SEQ * 1024 * 4);
  float*  attn0 = (float*) (ws + off); off = align256(off + 1024 * 4);
  float*  scor  = (float*) (ws + off); off = align256(off + (size_t)T_SEQ * 4);
  float*  stepf = (float*) (ws + off); off = align256(off + (size_t)S_SEG * 1024 * 4);
  __bf16* sbf   = (__bf16*)(ws + off); off = align256(off + (size_t)S_SEG * 1024 * 2);
  __bf16* wbfe  = (__bf16*)(ws + off); off = align256(off + (size_t)2 * G4E * 1024 * 2);
  float*  gxe   = (float*) (ws + off); off = align256(off + (size_t)S_SEG * 2048 * 4);
  float*  ent   = (float*) (ws + off); off = align256(off + (size_t)S_SEG * 512 * 4);
  float*  hglob = (float*) (ws + off); off = align256(off + (size_t)4 * HD * 4);
  int*    bar   = (int*)   (ws + off); off = align256(off + 64);
  (void)ws_size; (void)n_in; (void)in_sizes; (void)out_size;

  // 1) precision conversion for WMMA operands
  {
    int n = T_SEQ * D_IN;
    f2bf_kernel<<<(n + 255) / 256, 256, 0, stream>>>(elmo_sent, xbf, n);
    n = G4 * D_IN;
    f2bf_kernel<<<(n + 255) / 256, 256, 0, stream>>>(Wih_f, wbf, n);
    f2bf_kernel<<<(n + 255) / 256, 256, 0, stream>>>(Wih_b, wbf + (size_t)G4 * D_IN, n);
    n = G4E * 1024;
    f2bf_kernel<<<(n + 255) / 256, 256, 0, stream>>>(Wih_ef, wbfe, n);
    f2bf_kernel<<<(n + 255) / 256, 256, 0, stream>>>(Wih_eb, wbfe + (size_t)G4E * 1024, n);
  }
  init_kernel<<<1, 1024, 0, stream>>>(h0_sent, hglob, bar);

  // 2) gx = x @ [Wih_f;Wih_b].T -> [8192, 4096]  (bf16 WMMA, 16x64 per wave)
  gemm_bf16_wmma<<<dim3((2 * G4) / 512, T_SEQ / 16), 256, 0, stream>>>(
      xbf, wbf, gx, T_SEQ, 2 * G4, D_IN, D_IN, D_IN, 2 * G4);

  // 3) sentence recurrence: 16 WGs/dir, TDM-staged LDS-resident Whh slices
  {
    size_t ldsb = (size_t)(ROWS_PW * HD + HD + THR_R) * sizeof(float); // ~259 KB
    lstm_sent_kernel<<<2 * NWG_DIR, THR_R, ldsb, stream>>>(
        gx, Whh_f, Whh_b, b_f, b_b, c0_sent, lstm, hglob, bar);
  }

  // 4) attention key for ingredient 0
  attn0_kernel<<<4, 256, 0, stream>>>(W_attn, b_attn, elmo_ings, attn0);

  // 5) per-segment softmax + weighted pooling
  seg_attn_kernel<<<S_SEG, 256, 0, stream>>>(lstm, attn0, indices, scor, stepf);

  // 6) entity input projection (WMMA) and recurrence
  {
    int n = S_SEG * 1024;
    f2bf_kernel<<<(n + 255) / 256, 256, 0, stream>>>(stepf, sbf, n);
  }
  gemm_bf16_wmma<<<dim3((2 * G4E) / 512, S_SEG / 16), 256, 0, stream>>>(
      sbf, wbfe, gxe, S_SEG, 2 * G4E, 1024, 1024, 1024, 2 * G4E);
  lstm_ent_kernel<<<2, 1024, 0, stream>>>(gxe, Whh_ef, Whh_eb, b_ef, b_eb,
                                          h0_ent, c0_ent, ent);

  // 7) logits [64, 2]
  logits_kernel<<<1, 128, 0, stream>>>(ent, W_log, b_log, out);
}
